// CausalAttention_30159260353194
// MI455X (gfx1250) — matrix operand order repaired and checked by reading, not compiled, not run on hardware
//
#include <hip/hip_runtime.h>
#include <hip/hip_bf16.h>

#define BATCH 4
#define SEQ   2048
#define DIM   2048
#define INV_SQRT_D 0.02209708691207961f  // 1/sqrt(2048)

typedef __bf16 v8bf  __attribute__((ext_vector_type(8)));
typedef __bf16 v16bf __attribute__((ext_vector_type(16)));
typedef float  v8f   __attribute__((ext_vector_type(8)));

#if __has_builtin(__builtin_amdgcn_tensor_load_to_lds)
#define HAVE_TDM 1
typedef unsigned int u32x4 __attribute__((ext_vector_type(4)));
typedef int          i32x4 __attribute__((ext_vector_type(4)));
typedef int          i32x8 __attribute__((ext_vector_type(8)));
#endif

__device__ __forceinline__ v16bf cat16(v8bf a, v8bf b) {
  return __builtin_shufflevector(a, b, 0,1,2,3,4,5,6,7,8,9,10,11,12,13,14,15);
}

// ---------------- convert x: f32 -> bf16 (row major [B*S][D]) ----------------
__global__ void __launch_bounds__(256) k_cvt_x(const float* __restrict__ x,
                                               __bf16* __restrict__ xb) {
  long long i = ((long long)blockIdx.x * 256 + threadIdx.x) * 8;
  float4 a = *(const float4*)(x + i);
  float4 b = *(const float4*)(x + i + 4);
  v8bf o;
  o[0]=(__bf16)a.x; o[1]=(__bf16)a.y; o[2]=(__bf16)a.z; o[3]=(__bf16)a.w;
  o[4]=(__bf16)b.x; o[5]=(__bf16)b.y; o[6]=(__bf16)b.z; o[7]=(__bf16)b.w;
  *(v8bf*)(xb + i) = o;
}

// ------ convert+transpose W: Wt[z][n][k] = (bf16)W[z][k][n], z in {q,k,v} ------
__global__ void __launch_bounds__(256) k_cvt_wt(const float* __restrict__ Wq,
                                                const float* __restrict__ Wk,
                                                const float* __restrict__ Wv,
                                                __bf16* __restrict__ wt) {
  const float* W = (blockIdx.z == 0) ? Wq : (blockIdx.z == 1) ? Wk : Wv;
  __bf16* out = wt + (size_t)blockIdx.z * DIM * DIM;
  int idx = blockIdx.x * 256 + threadIdx.x;       // coalesced read along n
  int k = idx / DIM, n = idx % DIM;
  out[(size_t)n * DIM + k] = (__bf16)W[idx];
}

// ---------------- QKV GEMM: [8192 x 2048] @ [2048 x 2048] per z --------------
// wave tile 32x64 (2 A-frags x 4 B-frags = 8 acc); block = 8 waves on M => 256x64.
// z==0 -> Q (scaled by 1/sqrt(d)), z==1 -> K, z==2 -> V stored TRANSPOSED [d][B*S]
__global__ void __launch_bounds__(256) k_qkv(const __bf16* __restrict__ xb,
                                             const __bf16* __restrict__ wt,
                                             __bf16* __restrict__ qb,
                                             __bf16* __restrict__ kb,
                                             __bf16* __restrict__ vt) {
  const int wave = threadIdx.x >> 5;
  const int lane = threadIdx.x & 31;
  const int hi = lane >> 4, l = lane & 15;
  const int m0 = blockIdx.y * 256 + wave * 32;   // row in [0, 8192)
  const int n0 = blockIdx.x * 64;
  const int z  = blockIdx.z;
  const __bf16* w = wt + (size_t)z * DIM * DIM;

  v8f acc[2][4] = {{v8f{}, v8f{}, v8f{}, v8f{}}, {v8f{}, v8f{}, v8f{}, v8f{}}};
  const __bf16* arow0 = xb + (size_t)(m0 + l) * DIM;
  const __bf16* arow1 = arow0 + (size_t)16 * DIM;

  for (int kk = 0; kk < DIM; kk += 32) {
    const v8bf* a0 = (const v8bf*)(arow0 + kk + 8 * hi);
    const v8bf* a1 = (const v8bf*)(arow1 + kk + 8 * hi);
    v16bf A0 = cat16(a0[0], a0[2]);              // k = kk + (i&7) + 8*(2g+hi)
    v16bf A1 = cat16(a1[0], a1[2]);
    #pragma unroll
    for (int s = 0; s < 4; ++s) {
      const __bf16* brow = w + (size_t)(n0 + 16 * s + l) * DIM + kk + 8 * hi;
      v16bf Bf = cat16(*(const v8bf*)brow, *(const v8bf*)(brow + 16));
      acc[0][s] = __builtin_amdgcn_wmma_f32_16x16x32_bf16(
          false, A0, false, Bf, (short)0, acc[0][s], false, false);
      acc[1][s] = __builtin_amdgcn_wmma_f32_16x16x32_bf16(
          false, A1, false, Bf, (short)0, acc[1][s], false, false);
    }
  }

  #pragma unroll
  for (int a = 0; a < 2; ++a) {
    const int mb = m0 + 16 * a;
    if (z == 0) {
      #pragma unroll
      for (int s = 0; s < 4; ++s)
        #pragma unroll
        for (int r = 0; r < 8; ++r)
          qb[(size_t)(mb + r + 8 * hi) * DIM + n0 + 16 * s + l] =
              (__bf16)(acc[a][s][r] * INV_SQRT_D);
    } else if (z == 1) {
      #pragma unroll
      for (int s = 0; s < 4; ++s)
        #pragma unroll
        for (int r = 0; r < 8; ++r)
          kb[(size_t)(mb + r + 8 * hi) * DIM + n0 + 16 * s + l] = (__bf16)acc[a][s][r];
    } else {
      #pragma unroll
      for (int s = 0; s < 4; ++s)
        #pragma unroll
        for (int r = 0; r < 8; ++r)
          vt[(size_t)(n0 + 16 * s + l) * (BATCH * SEQ) + (mb + r + 8 * hi)] =
              (__bf16)acc[a][s][r];
    }
  }
}

// ------------- scores + causal softmax, one 16-query tile per block ----------
// Full 16x2048 f32 score row lives in LDS (single-pass softmax, no rescaling).
// Q tile staged to LDS via the Tensor Data Mover when available.
#define SROW (SEQ + 8)
__global__ void __launch_bounds__(256) k_scores(const __bf16* __restrict__ qb,
                                                const __bf16* __restrict__ kb,
                                                __bf16* __restrict__ pb) {
  __shared__ float  s_s[16 * SROW];      // 131584 B
  __shared__ __bf16 s_q[16 * DIM];       //  65536 B

  const int qt = blockIdx.x;             // 0..511
  const int batch = qt >> 7;
  const int t = qt & 127;                // query tile within batch
  const int q0 = t * 16;
  const int tid = threadIdx.x;
  const int wave = tid >> 5, lane = tid & 31, hi = lane >> 4, l = lane & 15;

  const __bf16* qbase = qb + (size_t)(batch * SEQ + q0) * DIM;

#ifdef HAVE_TDM
  if (wave == 0) {
    // Tensor DMA: 16 x 2048 bf16 tile, row-major, global -> LDS (D# per ISA 8.3/8.4)
    unsigned long long ga = (unsigned long long)(size_t)qbase;
    unsigned lds = (unsigned)(size_t)(void*)s_q;
    u32x4 g0;
    g0[0] = 1u;                                        // count=1 (valid descriptor)
    g0[1] = lds;                                       // lds_addr
    g0[2] = (unsigned)(ga & 0xFFFFFFFFu);              // global_addr[31:0]
    g0[3] = (unsigned)((ga >> 32) & 0x01FFFFFFu) | (2u << 30);  // addr[56:32] | type=2
    i32x8 g1;
    g1[0] = 0x00010000;            // wg_mask=0, data_size=1 (2 bytes)
    g1[1] = (int)(2048u << 16);    // tensor_dim0[15:0]=2048 in [31:16]
    g1[2] = (int)(16u << 16);      // tensor_dim0 hi=0 | tensor_dim1[15:0]=16
    g1[3] = (int)(2048u << 16);    // tensor_dim1 hi=0 | tile_dim0=2048
    g1[4] = 16;                    // tile_dim1=16, tile_dim2=0
    g1[5] = 2048;                  // tensor_dim0_stride[31:0]
    g1[6] = 0;                     // stride hi | tensor_dim1_stride lo
    g1[7] = 0;
    i32x4 z4 = {0, 0, 0, 0};
#if __clang_major__ >= 23
    i32x8 z8 = {0, 0, 0, 0, 0, 0, 0, 0};
    __builtin_amdgcn_tensor_load_to_lds(g0, g1, z4, z4, z8, 0);
#else
    __builtin_amdgcn_tensor_load_to_lds(g0, g1, z4, z4, 0);
#endif
    __builtin_amdgcn_s_wait_tensorcnt(0);
  }
#else
  for (int c = tid; c < 16 * DIM * 2 / 16; c += 256)
    ((uint4*)s_q)[c] = ((const uint4*)qbase)[c];
#endif
  __syncthreads();

  // causal: key tiles 0..t, processed as pairs so one A-frag feeds two WMMAs
  const int ktiles = t + 1;
  const int ktp = (ktiles + 1) >> 1;
  for (int p = wave; p < ktp; p += 8) {
    v8f acc0 = v8f{}, acc1 = v8f{};
    const __bf16* krow0 = kb + (size_t)(batch * SEQ + p * 32 + l) * DIM;
    const __bf16* krow1 = krow0 + (size_t)16 * DIM;
    for (int kk = 0; kk < DIM; kk += 32) {
      const v8bf* ap = (const v8bf*)(&s_q[l * DIM + kk + 8 * hi]);
      v16bf A = cat16(ap[0], ap[2]);
      const __bf16* b0 = krow0 + kk + 8 * hi;
      const __bf16* b1 = krow1 + kk + 8 * hi;
      v16bf B0 = cat16(*(const v8bf*)b0, *(const v8bf*)(b0 + 16));
      v16bf B1 = cat16(*(const v8bf*)b1, *(const v8bf*)(b1 + 16));
      acc0 = __builtin_amdgcn_wmma_f32_16x16x32_bf16(
          false, A, false, B0, (short)0, acc0, false, false);
      acc1 = __builtin_amdgcn_wmma_f32_16x16x32_bf16(
          false, A, false, B1, (short)0, acc1, false, false);
    }
    #pragma unroll
    for (int r = 0; r < 8; ++r) {
      s_s[(r + 8 * hi) * SROW + p * 32 + l]      = acc0[r];
      s_s[(r + 8 * hi) * SROW + p * 32 + 16 + l] = acc1[r];
    }
  }
  __syncthreads();

  // softmax: wave w owns rows 2w (lanes 0-15) and 2w+1 (lanes 16-31)
  const int row = wave * 2 + hi;
  const int q = q0 + row;                // global query index within batch
  float mx = -__builtin_inff();
  for (int j = l; j <= q; j += 16)
    mx = fmaxf(mx, s_s[row * SROW + j]);
  #pragma unroll
  for (int m = 1; m < 16; m <<= 1)
    mx = fmaxf(mx, __shfl_xor(mx, m, 16));

  float sum = 0.f;
  for (int j = l; j <= q; j += 16)
    sum += __expf(s_s[row * SROW + j] - mx);
  #pragma unroll
  for (int m = 1; m < 16; m <<= 1)
    sum += __shfl_xor(sum, m, 16);
  const float inv = 1.f / sum;

  // write probs; zero-pad to an even number of key tiles (= m0+32 bound in k_av)
  const int npad = ((ktiles + 1) & ~1) * 16;
  __bf16* prow = pb + (size_t)(batch * SEQ + q) * SEQ;
  for (int j = l; j < npad; j += 16) {
    float p = (j <= q) ? __expf(s_s[row * SROW + j] - mx) * inv : 0.f;
    prow[j] = (__bf16)p;
  }
}

// --------------------- O = P @ V (causal-bounded K loop) ---------------------
// wave tile 32x64; m0 is a multiple of 32 so both 16-row sub-tiles share the
// zero-padded bound kmax = m0+32 written by k_scores.
__global__ void __launch_bounds__(256) k_av(const __bf16* __restrict__ pb,
                                            const __bf16* __restrict__ vt,
                                            float* __restrict__ out) {
  const int wave = threadIdx.x >> 5;
  const int lane = threadIdx.x & 31;
  const int hi = lane >> 4, l = lane & 15;
  const int batch = blockIdx.z;
  const int m0 = blockIdx.y * 256 + wave * 32;   // query in [0, 2048)
  const int n0 = blockIdx.x * 64;
  const int kmax = m0 + 32;

  v8f acc[2][4] = {{v8f{}, v8f{}, v8f{}, v8f{}}, {v8f{}, v8f{}, v8f{}, v8f{}}};
  const __bf16* arow0 = pb + (size_t)(batch * SEQ + m0 + l) * SEQ;
  const __bf16* arow1 = arow0 + (size_t)16 * SEQ;

  for (int kk = 0; kk < kmax; kk += 32) {
    const v8bf* a0 = (const v8bf*)(arow0 + kk + 8 * hi);
    const v8bf* a1 = (const v8bf*)(arow1 + kk + 8 * hi);
    v16bf A0 = cat16(a0[0], a0[2]);
    v16bf A1 = cat16(a1[0], a1[2]);
    #pragma unroll
    for (int s = 0; s < 4; ++s) {
      const __bf16* brow = vt + (size_t)(n0 + 16 * s + l) * (BATCH * SEQ)
                              + batch * SEQ + kk + 8 * hi;
      v16bf Bf = cat16(*(const v8bf*)brow, *(const v8bf*)(brow + 16));
      acc[0][s] = __builtin_amdgcn_wmma_f32_16x16x32_bf16(
          false, A0, false, Bf, (short)0, acc[0][s], false, false);
      acc[1][s] = __builtin_amdgcn_wmma_f32_16x16x32_bf16(
          false, A1, false, Bf, (short)0, acc[1][s], false, false);
    }
  }
  #pragma unroll
  for (int a = 0; a < 2; ++a)
    #pragma unroll
    for (int s = 0; s < 4; ++s)
      #pragma unroll
      for (int r = 0; r < 8; ++r)
        out[(size_t)(batch * SEQ + m0 + 16 * a + r + 8 * hi) * DIM + n0 + 16 * s + l] =
            acc[a][s][r];
}

extern "C" void kernel_launch(void* const* d_in, const int* in_sizes, int n_in,
                              void* d_out, int out_size, void* d_ws, size_t ws_size,
                              hipStream_t stream) {
  const float* x  = (const float*)d_in[0];
  const float* Wq = (const float*)d_in[1];
  const float* Wk = (const float*)d_in[2];
  const float* Wv = (const float*)d_in[3];
  float* out = (float*)d_out;

  char* ws = (char*)d_ws;
  const size_t SZ_XB = (size_t)BATCH * SEQ * DIM * 2;   // 32 MB
  const size_t SZ_W  = (size_t)3 * DIM * DIM * 2;       // 24 MB
  __bf16* xb = (__bf16*)(ws);
  __bf16* wt = (__bf16*)(ws + SZ_XB);
  __bf16* qb = (__bf16*)(ws + SZ_XB + SZ_W);
  __bf16* kb = (__bf16*)(ws + SZ_XB + SZ_W + SZ_XB);
  __bf16* vt = (__bf16*)(ws + SZ_XB + SZ_W + 2 * SZ_XB);
  __bf16* pb = (__bf16*)(ws);                           // alias xb (dead by then)

  k_cvt_x <<<dim3((BATCH * SEQ * (size_t)DIM) / (256 * 8)), 256, 0, stream>>>(x, xb);
  k_cvt_wt<<<dim3(DIM * DIM / 256, 1, 3), 256, 0, stream>>>(Wq, Wk, Wv, wt);
  k_qkv   <<<dim3(DIM / 64, BATCH * SEQ / 256, 3), 256, 0, stream>>>(xb, wt, qb, kb, vt);
  k_scores<<<dim3(BATCH * SEQ / 16), 256, 0, stream>>>(qb, kb, pb);
  k_av    <<<dim3(DIM / 64, SEQ / 256, BATCH), 256, 0, stream>>>(pb, vt, out);
}
